// MultiheadAttention_77962246357117
// MI455X (gfx1250) — matrix operand order repaired and checked by reading, not compiled, not run on hardware
//
#include <hip/hip_runtime.h>
#include <hip/hip_bf16.h>

// ---------------------------------------------------------------------------
// MultiheadAttention forward for MI455X (gfx1250), bf16 WMMA pipeline
// with TDM / async global->LDS staging.
//   B=8, S=1024, D=1024, n_chunks(H)=16, head_dim=64
//   d_out = [ out: 8*1024*1024 f32 | weights: 16*8*1024*1024 f32 ]
// ---------------------------------------------------------------------------

typedef __attribute__((ext_vector_type(16))) __bf16 v16bf;
typedef __attribute__((ext_vector_type(8)))  float  v8f;
typedef __attribute__((ext_vector_type(4)))  unsigned int u32x4;
typedef __attribute__((ext_vector_type(8)))  int          i32x8;
typedef __attribute__((ext_vector_type(4)))  int          i32x4;
typedef __attribute__((ext_vector_type(4)))  int          v4i;

#define D_MODEL 1024
#define NHEADS  16
#define DHEAD   64
#define BATCH   8
#define SEQ     1024
#define ROWS    (BATCH * SEQ)   // 8192

// ---- feature guards --------------------------------------------------------
#if __has_builtin(__builtin_amdgcn_tensor_load_to_lds) && \
    __has_builtin(__builtin_amdgcn_s_wait_tensorcnt)
#define HAVE_TDM 1
#else
#define HAVE_TDM 0
#endif

#if __has_builtin(__builtin_amdgcn_global_load_async_to_lds_b128)
#define HAVE_ASYNC 1
#else
#define HAVE_ASYNC 0
#endif

#if __has_builtin(__builtin_amdgcn_s_wait_asynccnt)
#define ASYNC_WAIT(n) __builtin_amdgcn_s_wait_asynccnt(n)
#elif HAVE_ASYNC
#define ASYNC_WAIT(n) asm volatile("s_wait_asynccnt %0" ::"i"(n) : "memory")
#else
#define ASYNC_WAIT(n)
#endif

typedef __attribute__((address_space(3))) char lds_char;
__device__ __forceinline__ unsigned lds_addr_of(void* p) {
  return (unsigned)(size_t)(lds_char*)p;
}

// ---- fp32 -> bf16 round-to-nearest-even -----------------------------------
__device__ __forceinline__ unsigned short f2bf(float x) {
  unsigned u = __float_as_uint(x);
  u += 0x7FFFu + ((u >> 16) & 1u);
  return (unsigned short)(u >> 16);
}
__device__ __forceinline__ unsigned f2bf_pack(float lo, float hi) {
  return (unsigned)f2bf(lo) | ((unsigned)f2bf(hi) << 16);
}

union FragU { v16bf v; unsigned u[8]; };

// ---- async global -> LDS 16B copy (per lane) -------------------------------
// Builtin expects (int4 addrspace(1)*, int4 addrspace(3)*, imm, imm) per the
// clang diagnostic ("int __device__ *" == LangAS::cuda_device == AS1).
__device__ __forceinline__ void async_copy_b128(void* lds_dst, const void* gsrc) {
#if HAVE_ASYNC
  __builtin_amdgcn_global_load_async_to_lds_b128(
      (__attribute__((address_space(1))) v4i*)(v4i*)(void*)gsrc,
      (__attribute__((address_space(3))) v4i*)(v4i*)lds_dst, 0, 0);
#else
  *(uint4*)lds_dst = *(const uint4*)gsrc;
#endif
}

// ---- TDM: 2-D bf16 tile (tile_d0 contiguous elems x tile_d1 rows) ----------
#if HAVE_TDM
__device__ __forceinline__ void tdm_load_tile_2d(unsigned lds_byte_addr,
                                                 const void* tile_gaddr,
                                                 unsigned tensor_d0,
                                                 unsigned tensor_d1,
                                                 unsigned tile_d0,
                                                 unsigned tile_d1,
                                                 unsigned long long stride0) {
  unsigned long long ga = (unsigned long long)tile_gaddr;
  u32x4 g0 = {
      1u,                                               // count=1, user desc
      lds_byte_addr,                                    // lds_addr [63:32]
      (unsigned)ga,                                     // global_addr lo
      (unsigned)((ga >> 32) & 0x01FFFFFFu) | (2u << 30) // addr hi | type=2
  };
  i32x8 g1 = {
      (int)(1u << 16),                                              // data_size=2B
      (int)((tensor_d0 & 0xFFFFu) << 16),                           // dim0 lo16
      (int)(((tensor_d0 >> 16) & 0xFFFFu) | ((tensor_d1 & 0xFFFFu) << 16)),
      (int)(((tensor_d1 >> 16) & 0xFFFFu) | ((tile_d0 & 0xFFFFu) << 16)),
      (int)(tile_d1 & 0xFFFFu),                                     // tile_d1, d2=0
      (int)(unsigned)(stride0 & 0xFFFFFFFFull),                     // stride0 lo32
      (int)(unsigned)((stride0 >> 32) & 0xFFFFull),                 // stride0 hi16
      0
  };
  i32x4 z = {0, 0, 0, 0};
#if defined(__clang_major__) && (__clang_major__ >= 23)
  i32x8 z8 = {0, 0, 0, 0, 0, 0, 0, 0};
  __builtin_amdgcn_tensor_load_to_lds(g0, g1, z, z, z8, 0);
#else
  __builtin_amdgcn_tensor_load_to_lds(g0, g1, z, z, 0);
#endif
}
#endif

// ---- WMMA fragment loaders (per CDNA5 ISA 7.12.2 VGPR layouts) -------------
// A: 16x32 bf16 (MxK). lane%16 -> M, lane/16 selects K-phase.
__device__ __forceinline__ v16bf load_a_bf16(const unsigned short* __restrict__ A,
                                             int lda, int m0, int k0) {
  int lane  = threadIdx.x & 31;
  int m     = m0 + (lane & 15);
  int half8 = (lane >> 4) * 8;
  const unsigned short* row = A + (size_t)m * lda + k0;
  FragU f;
#pragma unroll
  for (int i = 0; i < 4; ++i)
    f.u[i] = *(const unsigned*)(row + half8 + 2 * i);
#pragma unroll
  for (int i = 0; i < 4; ++i)
    f.u[4 + i] = *(const unsigned*)(row + 16 + half8 + 2 * i);
  return f.v;
}

// A fragment built on the fly from an fp32 row-major matrix.
__device__ __forceinline__ v16bf load_a_f32(const float* __restrict__ A,
                                            int lda, int m0, int k0) {
  int lane  = threadIdx.x & 31;
  int m     = m0 + (lane & 15);
  int half8 = (lane >> 4) * 8;
  const float* row = A + (size_t)m * lda + k0;
  FragU f;
#pragma unroll
  for (int i = 0; i < 4; ++i) {
    float2 p = *(const float2*)(row + half8 + 2 * i);
    f.u[i] = f2bf_pack(p.x, p.y);
  }
#pragma unroll
  for (int i = 0; i < 4; ++i) {
    float2 p = *(const float2*)(row + 16 + half8 + 2 * i);
    f.u[4 + i] = f2bf_pack(p.x, p.y);
  }
  return f.v;
}

// B: 32x16 bf16 (KxN) from B^T row-major global memory (row=n, col=k).
__device__ __forceinline__ v16bf load_bt_bf16(const unsigned short* __restrict__ Bt,
                                              int ldb, int n0, int k0) {
  int lane = threadIdx.x & 31;
  int n    = n0 + (lane & 15);
  int kh   = (lane >> 4) * 8;
  const unsigned short* row = Bt + (size_t)n * ldb + k0 + kh;
  FragU f;
#pragma unroll
  for (int i = 0; i < 8; ++i)
    f.u[i] = *(const unsigned*)(row + (i >> 2) * 16 + 2 * (i & 3));
  return f.v;
}

// B fragment from an LDS tile with row stride ldt (bf16 elements).
__device__ __forceinline__ v16bf load_bt_lds(const unsigned short* tile, int ldt,
                                             int n0, int k0) {
  int lane = threadIdx.x & 31;
  const unsigned short* row =
      tile + (n0 + (lane & 15)) * ldt + k0 + (lane >> 4) * 8;
  FragU f;
#pragma unroll
  for (int i = 0; i < 8; ++i)
    f.u[i] = *(const unsigned*)(row + (i >> 2) * 16 + 2 * (i & 3));
  return f.v;
}

__device__ __forceinline__ v8f wmma_bf16(v16bf a, v16bf b, v8f c) {
  return __builtin_amdgcn_wmma_f32_16x16x32_bf16(false, a, false, b,
                                                 (short)0, c, false, false);
}

// ---------------------------------------------------------------------------
// Kernel 1: fp32 -> bf16 bulk convert (n multiple of 4)
// ---------------------------------------------------------------------------
__global__ void cvt_f32_bf16(const float* __restrict__ src,
                             unsigned short* __restrict__ dst, int n) {
  int stride = gridDim.x * blockDim.x * 4;
  for (int i = (blockIdx.x * blockDim.x + threadIdx.x) * 4; i < n; i += stride) {
    float4 v = *(const float4*)(src + i);
    *(unsigned*)(dst + i)     = f2bf_pack(v.x, v.y);
    *(unsigned*)(dst + i + 2) = f2bf_pack(v.z, v.w);
  }
}

// ---------------------------------------------------------------------------
// Kernel 2: C[M,N] = A[M,K] * Bt[N,K]^T   (bf16 in, f32 accum)
// Block tile 128x64, 8 waves of 32x32. B tile (64x32) staged in LDS by the
// Tensor Data Mover (double-buffered, s_wait_tensorcnt), A direct from L2.
// ---------------------------------------------------------------------------
template <bool F32_OUT>
__global__ void gemm_rowrow(const unsigned short* __restrict__ A,
                            const unsigned short* __restrict__ Bt,
                            void* __restrict__ C, int N, int K) {
  __shared__ unsigned short btile[2][64 * 32];   // 2 x 4KB
  int w      = threadIdx.x >> 5;
  int lane   = threadIdx.x & 31;
  int mBase  = blockIdx.x * 128 + (w & 3) * 32;
  int nBlock = blockIdx.y * 64;
  int nOff   = (w >> 2) * 32;
  v8f acc[2][2] = {};

  auto stage = [&](int k0, int bufi) {
#if HAVE_TDM
    if (threadIdx.x < 32)
      tdm_load_tile_2d(lds_addr_of(&btile[bufi][0]),
                       Bt + (size_t)nBlock * K + k0,
                       (unsigned)K, 1024u, 32u, 64u, (unsigned long long)K);
#else
    int t = threadIdx.x;            // 256 threads x 16B = 4KB tile
    int row = t >> 2, ch = (t & 3) * 8;
    *(uint4*)(&btile[bufi][row * 32 + ch]) =
        *(const uint4*)(Bt + (size_t)(nBlock + row) * K + k0 + ch);
#endif
  };

  stage(0, 0);
  int steps = K / 32;
  for (int kk = 0; kk < steps; ++kk) {
    int k0 = kk * 32;
    if (kk + 1 < steps) stage(k0 + 32, (kk + 1) & 1);
#if HAVE_TDM
    if (threadIdx.x < 32) {
      if (kk + 1 < steps) __builtin_amdgcn_s_wait_tensorcnt(1);
      else                __builtin_amdgcn_s_wait_tensorcnt(0);
    }
#endif
    __syncthreads();                       // tile kk ready for all waves
    const unsigned short* tile = &btile[kk & 1][0];
    if (k0 + 32 < K)
      __builtin_prefetch(A + (size_t)(mBase + lane) * K + k0 + 32, 0, 1);
    v16bf a0 = load_a_bf16(A, K, mBase, k0);
    v16bf a1 = load_a_bf16(A, K, mBase + 16, k0);
    v16bf b0 = load_bt_lds(tile, 32, nOff, 0);
    v16bf b1 = load_bt_lds(tile, 32, nOff + 16, 0);
    acc[0][0] = wmma_bf16(a0, b0, acc[0][0]);
    acc[0][1] = wmma_bf16(a0, b1, acc[0][1]);
    acc[1][0] = wmma_bf16(a1, b0, acc[1][0]);
    acc[1][1] = wmma_bf16(a1, b1, acc[1][1]);
    __syncthreads();                       // everyone done reading tile kk
  }

  int half = lane >> 4, nCol = lane & 15;
#pragma unroll
  for (int im = 0; im < 2; ++im)
#pragma unroll
    for (int jn = 0; jn < 2; ++jn)
#pragma unroll
      for (int v = 0; v < 8; ++v) {
        int   m   = mBase + im * 16 + half * 8 + v;
        int   n   = nBlock + nOff + jn * 16 + nCol;
        float val = acc[im][jn][v];
        if (F32_OUT) ((float*)C)[(size_t)m * N + n] = val;
        else         ((unsigned short*)C)[(size_t)m * N + n] = f2bf(val);
      }
}

// ---------------------------------------------------------------------------
// Kernel 3: scores (Q·K^T / 8) for one (h,b,32-row block), softmax in LDS,
// normalized weights streamed to d_out.
//   - K tiles (64 t-rows x 64 head-dims, 8KB) staged via async global->LDS,
//     double-buffered with s_wait_asynccnt.
//   - scores kept in 32 x 1032 f32 dynamic LDS (132KB).
// ---------------------------------------------------------------------------
__global__ void scores_softmax(const unsigned short* __restrict__ Qp,
                               const unsigned short* __restrict__ Kp,
                               float* __restrict__ Wout) {
  extern __shared__ float sc[];                    // 32 x 1032 f32
  __shared__ unsigned short kt[2][64 * 64];        // 2 x 8KB K tiles
  const int LDS_LD = 1032;
  int hb = blockIdx.x;                 // h*8 + b
  int h  = hb >> 3, b = hb & 7;
  int s0 = blockIdx.y * 32;
  int w    = threadIdx.x >> 5;
  int lane = threadIdx.x & 31;
  int mT = (w & 1) * 16;               // 2 waves along M (32 rows)
  int nT = (w >> 1) * 16;              // 4 waves along N within a 64-col t-tile
  const unsigned short* Qbase = Qp + (size_t)b * SEQ * D_MODEL + h * DHEAD;
  const unsigned short* Kbase = Kp + (size_t)b * SEQ * D_MODEL + h * DHEAD;

  // Q fragments for this wave's 16 rows (head dim 64 = two k-steps), hoisted.
  v16bf a0 = load_a_bf16(Qbase, D_MODEL, s0 + mT, 0);
  v16bf a1 = load_a_bf16(Qbase, D_MODEL, s0 + mT, 32);
  int half = lane >> 4, nCol = lane & 15;

  auto stageK = [&](int tt, int bufi) {
#pragma unroll
    for (int j = 0; j < 2; ++j) {                  // 512 x 16B chunks
      int ch  = threadIdx.x + j * 256;
      int row = ch >> 3;                           // 64 rows
      int col = (ch & 7) * 8;                      // 8 bf16 = 16B
      async_copy_b128(&kt[bufi][row * 64 + col],
                      Kbase + (size_t)(tt * 64 + row) * D_MODEL + col);
    }
  };

  stageK(0, 0);
  for (int tt = 0; tt < SEQ / 64; ++tt) {
    if (tt + 1 < SEQ / 64) { stageK(tt + 1, (tt + 1) & 1); ASYNC_WAIT(2); }
    else                   { ASYNC_WAIT(0); }
    __syncthreads();                               // tile tt landed in LDS
    const unsigned short* tile = &kt[tt & 1][0];
    v8f acc = {};
    acc = wmma_bf16(a0, load_bt_lds(tile, 64, nT, 0),  acc);
    acc = wmma_bf16(a1, load_bt_lds(tile, 64, nT, 32), acc);
    int col = tt * 64 + nT + nCol;
#pragma unroll
    for (int v = 0; v < 8; ++v) {
      int m = mT + half * 8 + v;
      sc[m * LDS_LD + col] = acc[v] * 0.125f;      // 1/sqrt(64)
    }
    __syncthreads();                               // done reading buf tt&1
  }

  // Softmax: 4 rows per wave, lanes stride the 1024 columns (float4).
  size_t outBase = ((size_t)hb * SEQ + s0) * SEQ;
  for (int r = 0; r < 4; ++r) {
    int m = w * 4 + r;
    float* row = sc + m * LDS_LD;
    float mx = -3.4e38f;
    for (int c = lane * 4; c < SEQ; c += 128) {
      float4 v = *(float4*)(row + c);
      mx = fmaxf(mx, fmaxf(fmaxf(v.x, v.y), fmaxf(v.z, v.w)));
    }
#pragma unroll
    for (int off = 16; off; off >>= 1) mx = fmaxf(mx, __shfl_xor(mx, off, 32));
    float sum = 0.f;
    for (int c = lane * 4; c < SEQ; c += 128) {
      float4 v = *(float4*)(row + c);
      v.x = __expf(v.x - mx); v.y = __expf(v.y - mx);
      v.z = __expf(v.z - mx); v.w = __expf(v.w - mx);
      *(float4*)(row + c) = v;
      sum += v.x + v.y + v.z + v.w;
    }
#pragma unroll
    for (int off = 16; off; off >>= 1) sum += __shfl_xor(sum, off, 32);
    float inv = 1.0f / sum;
    float* orow = Wout + outBase + (size_t)m * SEQ;
    for (int c = lane * 4; c < SEQ; c += 128) {
      float4 v = *(float4*)(row + c);
      v.x *= inv; v.y *= inv; v.z *= inv; v.w *= inv;
      *(float4*)(orow + c) = v;
    }
  }
}

// ---------------------------------------------------------------------------
// Kernel 4: transpose V per (h,b): Vt[hb][d][t] = Vp[b][t][h*64+d]   (bf16)
// ---------------------------------------------------------------------------
__global__ void vtrans(const unsigned short* __restrict__ Vp,
                       unsigned short* __restrict__ Vt) {
  int total = NHEADS * BATCH * DHEAD * SEQ;
  for (int idx = blockIdx.x * blockDim.x + threadIdx.x; idx < total;
       idx += gridDim.x * blockDim.x) {
    int t  = idx & (SEQ - 1);
    int d  = (idx >> 10) & (DHEAD - 1);
    int hb = idx >> 16;                   // 64*1024 per (h,b)
    int b = hb & 7, h = hb >> 3;
    Vt[idx] = Vp[(size_t)(b * SEQ + t) * D_MODEL + h * DHEAD + d];
  }
}

// ---------------------------------------------------------------------------
// Kernel 5: attended = weights(f32) x V, scattered into the permute-quirk
// layout X (bf16 [8192 x 1024]) so the output projection is a plain GEMM.
//   X flat index = (h<<19) + (s<<9) + (b<<6) + d
// ---------------------------------------------------------------------------
__global__ void attn_av(const float* __restrict__ Wts,         // [hb][s][t]
                        const unsigned short* __restrict__ Vt, // [hb][d][t]
                        unsigned short* __restrict__ X) {
  int hb = blockIdx.x;
  int h  = hb >> 3, b = hb & 7;
  int w    = threadIdx.x >> 5;
  int lane = threadIdx.x & 31;
  int m0 = blockIdx.y * 128 + w * 16;    // s rows (16 per wave)
  const float*          Wbase = Wts + (size_t)hb * SEQ * SEQ;
  const unsigned short* Vbase = Vt + (size_t)hb * DHEAD * SEQ;
  v8f acc[4] = {};
  for (int k0 = 0; k0 < SEQ; k0 += 32) {
    v16bf a = load_a_f32(Wbase, SEQ, m0, k0);  // fp32 weights -> bf16 frag
#pragma unroll
    for (int nt = 0; nt < 4; ++nt) {
      v16bf bf = load_bt_bf16(Vbase, SEQ, nt * 16, k0);
      acc[nt] = wmma_bf16(a, bf, acc[nt]);
    }
  }
  int half = lane >> 4, nCol = lane & 15;
#pragma unroll
  for (int nt = 0; nt < 4; ++nt)
#pragma unroll
    for (int v = 0; v < 8; ++v) {
      int s = m0 + half * 8 + v;
      int d = nt * 16 + nCol;
      size_t xi = ((size_t)h << 19) + ((size_t)s << 9) + ((size_t)b << 6) + d;
      X[xi] = f2bf(acc[nt][v]);
    }
}

// ---------------------------------------------------------------------------
extern "C" void kernel_launch(void* const* d_in, const int* in_sizes, int n_in,
                              void* d_out, int out_size, void* d_ws, size_t ws_size,
                              hipStream_t stream) {
  const float* Q  = (const float*)d_in[0];
  const float* K  = (const float*)d_in[1];
  const float* V  = (const float*)d_in[2];
  const float* Wq = (const float*)d_in[3];
  const float* Wk = (const float*)d_in[4];
  const float* Wv = (const float*)d_in[5];
  const float* Wo = (const float*)d_in[6];

  const size_t QKV = (size_t)ROWS * D_MODEL;      // 8388608
  const size_t WEL = (size_t)D_MODEL * D_MODEL;   // 1048576

  unsigned short* ws = (unsigned short*)d_ws;
  size_t off = 0;
  unsigned short* Qb  = ws + off; off += QKV;
  unsigned short* Kb  = ws + off; off += QKV;
  unsigned short* Vb  = ws + off; off += QKV;
  unsigned short* Wqb = ws + off; off += WEL;
  unsigned short* Wkb = ws + off; off += WEL;
  unsigned short* Wvb = ws + off; off += WEL;
  unsigned short* Wob = ws + off; off += WEL;
  unsigned short* Qp  = ws + off; off += QKV;
  unsigned short* Kp  = ws + off; off += QKV;
  unsigned short* Vp  = ws + off; off += QKV;
  unsigned short* Vt  = ws + off; off += QKV;
  unsigned short* X   = ws + off; off += QKV;

  float* outp = (float*)d_out;
  float* wts  = outp + (size_t)BATCH * SEQ * D_MODEL;  // weights region

  dim3 blk(256);

  // 1) fp32 -> bf16
  cvt_f32_bf16<<<2048, blk, 0, stream>>>(Q,  Qb,  (int)QKV);
  cvt_f32_bf16<<<2048, blk, 0, stream>>>(K,  Kb,  (int)QKV);
  cvt_f32_bf16<<<2048, blk, 0, stream>>>(V,  Vb,  (int)QKV);
  cvt_f32_bf16<<<512,  blk, 0, stream>>>(Wq, Wqb, (int)WEL);
  cvt_f32_bf16<<<512,  blk, 0, stream>>>(Wk, Wkb, (int)WEL);
  cvt_f32_bf16<<<512,  blk, 0, stream>>>(Wv, Wvb, (int)WEL);
  cvt_f32_bf16<<<512,  blk, 0, stream>>>(Wo, Wob, (int)WEL);

  // 2) projections: Qp = Q Wq^T, etc. (TDM-staged B tiles)
  dim3 g1(ROWS / 128, D_MODEL / 64);
  gemm_rowrow<false><<<g1, blk, 0, stream>>>(Qb, Wqb, Qp, D_MODEL, D_MODEL);
  gemm_rowrow<false><<<g1, blk, 0, stream>>>(Kb, Wkb, Kp, D_MODEL, D_MODEL);
  gemm_rowrow<false><<<g1, blk, 0, stream>>>(Vb, Wvb, Vp, D_MODEL, D_MODEL);

  // 3) scores + softmax -> weights output (async-staged K tiles)
  dim3 g2(NHEADS * BATCH, SEQ / 32);
  scores_softmax<<<g2, blk, 32 * 1032 * sizeof(float), stream>>>(Qp, Kp, wts);

  // 4) V transpose for contiguous-k WMMA B fragments
  vtrans<<<4096, blk, 0, stream>>>(Vp, Vt);

  // 5) attended = weights x V, scattered into permute-quirk layout
  dim3 g3(NHEADS * BATCH, SEQ / 128);
  attn_av<<<g3, blk, 0, stream>>>(wts, Vt, X);

  // 6) out = X Wo^T  (fp32 stores straight to d_out)
  gemm_rowrow<true><<<g1, blk, 0, stream>>>(X, Wob, outp, D_MODEL, D_MODEL);
}
